// LatentSDE_4423816315315
// MI455X (gfx1250) — compile-verified, matrix operand order checked by reading, not run on hardware
//
#include <hip/hip_runtime.h>

typedef __bf16 bf16;
typedef __attribute__((ext_vector_type(16))) bf16  v16bf;
typedef __attribute__((ext_vector_type(8)))  float v8f;

#define T_  256
#define B_  1024
#define D_  16
#define L_  32
#define C_  64
#define H_  256

// ---------------- WMMA helpers (CDNA5 v_wmma_f32_16x16x32_bf16) ----------------

__device__ __forceinline__ v8f wmma_bf(v16bf a, v16bf b, v8f c) {
  // 8 args: (neg_a, A, neg_b, B, c_mod, C, reuse_a, reuse_b)
  return __builtin_amdgcn_wmma_f32_16x16x32_bf16(false, a, false, b, (short)0, c,
                                                 false, false);
}

// A fragment: 16x32 bf16 tile from row-major [.., lda] at (row0, k0).
// ISA layout: lanes 0-15 M=lane kb=0; lanes 16-31 M=lane-16 kb=8;
// element e -> K = kb + e + (e&8)
__device__ __forceinline__ v16bf frag_a(const bf16* p, int lda, int row0, int k0,
                                        int kmax) {
  int lane = threadIdx.x & 31;
  int m  = lane & 15;
  int kb = (lane & 16) ? 8 : 0;
  const bf16* r = p + (size_t)(row0 + m) * lda + k0;
  v16bf out;
#pragma unroll
  for (int e = 0; e < 16; ++e) {
    int ko = kb + e + (e & 8);
    out[e] = (ko < kmax) ? r[ko] : (bf16)0.f;
  }
  return out;
}

// B fragment: 32x16 (KxN) where B[k][n] = W[n][k], W row-major [Ntot, ldw].
// Mirrored layout: lane = N column, same K striping as A.
__device__ __forceinline__ v16bf frag_b(const bf16* W, int ldw, int k0, int n0,
                                        int kmax) {
  int lane = threadIdx.x & 31;
  int n  = lane & 15;
  int kb = (lane & 16) ? 8 : 0;
  const bf16* r = W + (size_t)(n0 + n) * ldw + k0;
  v16bf out;
#pragma unroll
  for (int e = 0; e < 16; ++e) {
    int ko = kb + e + (e & 8);
    out[e] = (ko < kmax) ? r[ko] : (bf16)0.f;
  }
  return out;
}

// ---------------- scalar math ----------------

__device__ __forceinline__ float sigf(float x) { return 1.f / (1.f + __expf(-x)); }
__device__ __forceinline__ float splusf(float x) {
  return (x > 20.f) ? x : log1pf(__expf(x));
}
__device__ __forceinline__ float wave_sum(float v) {
#pragma unroll
  for (int m = 16; m; m >>= 1) v += __shfl_xor(v, m, 32);
  return v;
}

// ---------------- utility kernels ----------------

__global__ void cvt_bf16_kernel(const float* __restrict__ s, bf16* __restrict__ d,
                                int n) {
  for (int i = blockIdx.x * blockDim.x + threadIdx.x; i < n;
       i += gridDim.x * blockDim.x)
    d[i] = (bf16)s[i];
}
__global__ void zero_f32_kernel(float* p, int n) {
  for (int i = blockIdx.x * blockDim.x + threadIdx.x; i < n;
       i += gridDim.x * blockDim.x)
    p[i] = 0.f;
}
__global__ void zero_bf16_kernel(bf16* p, int n) {
  for (int i = blockIdx.x * blockDim.x + threadIdx.x; i < n;
       i += gridDim.x * blockDim.x)
    p[i] = (bf16)0.f;
}

// ---------------- GRU step: h_new = GRUCell(h, x_t) ----------------
// grid (B/64, H/64), 128 threads (4 waves); wave handles 16 rows x 64 cols.
__global__ void gru_step_kernel(const bf16* __restrict__ xt,      // [B,D]
                                const bf16* __restrict__ h_bf,    // [B,H]
                                const float* __restrict__ h_f,    // [B,H]
                                float* __restrict__ hn_f, bf16* __restrict__ hn_bf,
                                const bf16* __restrict__ Wih,     // [3H,D]
                                const bf16* __restrict__ Whh,     // [3H,H]
                                const float* __restrict__ bih,
                                const float* __restrict__ bhh) {
  int lane = threadIdx.x & 31, w = threadIdx.x >> 5;
  int row0 = blockIdx.x * 64 + w * 16;
  int cb   = blockIdx.y * 64;

  for (int nt = 0; nt < 4; ++nt) {
    int n0 = cb + nt * 16;
    v8f ai[3] = {};
    v8f ah[3] = {};
    // input-gate GEMM, K = D = 16 (zero-padded chunk)
    v16bf ax = frag_a(xt, D_, row0, 0, D_);
#pragma unroll
    for (int g = 0; g < 3; ++g) {
      v16bf bx = frag_b(Wih, D_, 0, g * H_ + n0, D_);
      ai[g] = wmma_bf(ax, bx, ai[g]);
    }
    // hidden-gate GEMM, K = H = 256
    for (int kc = 0; kc < H_ / 32; ++kc) {
      v16bf ahf = frag_a(h_bf, H_, row0, kc * 32, 32);
#pragma unroll
      for (int g = 0; g < 3; ++g) {
        v16bf bh = frag_b(Whh, H_, kc * 32, g * H_ + n0, 32);
        ah[g] = wmma_bf(ahf, bh, ah[g]);
      }
    }
    int n = lane & 15, mb = (lane & 16) ? 8 : 0;
    int col = n0 + n;
    float bi0 = bih[col], bi1 = bih[H_ + col], bi2 = bih[2 * H_ + col];
    float bh0 = bhh[col], bh1 = bhh[H_ + col], bh2 = bhh[2 * H_ + col];
#pragma unroll
    for (int v = 0; v < 8; ++v) {
      int m = row0 + mb + v;
      float r  = sigf(ai[0][v] + bi0 + ah[0][v] + bh0);
      float zg = sigf(ai[1][v] + bi1 + ah[1][v] + bh1);
      float nn = tanhf(ai[2][v] + bi2 + r * (ah[2][v] + bh2));
      float hold = h_f[(size_t)m * H_ + col];
      float hnew = (1.f - zg) * nn + zg * hold;
      hn_f [(size_t)m * H_ + col] = hnew;
      hn_bf[(size_t)m * H_ + col] = (bf16)hnew;
    }
  }
}

// ---------------- ctx head: ctx[t] = h_new @ enc_W^T + enc_b ----------------
// grid B/64, 128 threads.
__global__ void ctx_step_kernel(const bf16* __restrict__ h_bf,
                                const bf16* __restrict__ encW,  // [C,H]
                                const float* __restrict__ enc_b,
                                bf16* __restrict__ ctx_t) {     // [B,C]
  int lane = threadIdx.x & 31, w = threadIdx.x >> 5;
  int row0 = blockIdx.x * 64 + w * 16;
  for (int nt = 0; nt < C_ / 16; ++nt) {
    v8f acc = {};
    for (int kc = 0; kc < H_ / 32; ++kc)
      acc = wmma_bf(frag_a(h_bf, H_, row0, kc * 32, 32),
                    frag_b(encW, H_, kc * 32, nt * 16, 32), acc);
    int n = lane & 15, mb = (lane & 16) ? 8 : 0;
    int col = nt * 16 + n;
    float b = enc_b[col];
#pragma unroll
    for (int v = 0; v < 8; ++v)
      ctx_t[(size_t)(row0 + mb + v) * C_ + col] = (bf16)(acc[v] + b);
  }
}

// ---------------- posterior init cond + KL + decode lp at t=0 ----------------
__global__ void qz0_kernel(const bf16* __restrict__ ctx0,       // [B,C]
                           const float* __restrict__ qz0_W,     // [2L,C]
                           const float* __restrict__ qz0_b,
                           const float* __restrict__ eps0,      // [B,L]
                           const float* __restrict__ pz0_mean,
                           const float* __restrict__ pz0_logstd,
                           float* __restrict__ zf0, bf16* __restrict__ zbf0,
                           const float* __restrict__ xs0,       // [B,D]
                           const float* __restrict__ projW,     // [D,L]
                           const float* __restrict__ proj_b,
                           const float* __restrict__ noise_std,
                           float* __restrict__ acc) {
  int b = blockIdx.x * blockDim.x + threadIdx.x;  // 1024 threads total
  int lane = threadIdx.x & 31;
  float cv[C_];
#pragma unroll 8
  for (int c = 0; c < C_; ++c) cv[c] = (float)ctx0[(size_t)b * C_ + c];
  float zrow[L_];
  float kl = 0.f;
  for (int l = 0; l < L_; ++l) {
    float qm = qz0_b[l], ql = qz0_b[L_ + l];
    for (int c = 0; c < C_; ++c) {
      qm += cv[c] * qz0_W[(size_t)l * C_ + c];
      ql += cv[c] * qz0_W[(size_t)(L_ + l) * C_ + c];
    }
    float z0 = qm + __expf(ql) * eps0[(size_t)b * L_ + l];
    zrow[l] = z0;
    zf0 [(size_t)b * L_ + l] = z0;
    zbf0[(size_t)b * L_ + l] = (bf16)z0;
    float pm = pz0_mean[l], pl = pz0_logstd[l];
    kl += pl - ql +
          (__expf(2.f * ql) + (qm - pm) * (qm - pm)) / (2.f * __expf(2.f * pl)) -
          0.5f;
  }
  // decode lp at t=0
  float ns = noise_std[0], lns = __logf(ns);
  float lp = 0.f;
  for (int d = 0; d < D_; ++d) {
    float x = proj_b[d];
#pragma unroll 8
    for (int l = 0; l < L_; ++l) x += zrow[l] * projW[d * L_ + l];
    float df = (xs0[(size_t)b * D_ + d] - x) / ns;
    lp += -0.5f * df * df - lns - 0.91893853320467274f;
  }
  kl = wave_sum(kl);
  lp = wave_sum(lp);
  if (lane == 0) {
    atomicAdd(&acc[1], kl);
    atomicAdd(&acc[0], lp);
  }
}

// ---------------- fused SDE Euler-Maruyama step ----------------
// grid B/32, 64 threads (2 waves); wave handles a 16-row batch slab.
__global__ void sde_step_kernel(
    int k, const float* __restrict__ ts,
    const float* __restrict__ zf_cur, const bf16* __restrict__ zbf_cur,
    float* __restrict__ zf_next, bf16* __restrict__ zbf_next,
    const bf16* __restrict__ ctx_row,                     // ctx + idx*B*C
    const bf16* __restrict__ fW1, const float* __restrict__ f_b1,
    const bf16* __restrict__ fW2, const float* __restrict__ f_b2,
    const bf16* __restrict__ fW3, const float* __restrict__ f_b3,
    const bf16* __restrict__ hW1, const float* __restrict__ h_b1,
    const bf16* __restrict__ hW2, const float* __restrict__ h_b2,
    const bf16* __restrict__ hW3, const float* __restrict__ h_b3,
    const float* __restrict__ gW1, const float* __restrict__ gb1,
    const float* __restrict__ gW2, const float* __restrict__ gb2,
    const float* __restrict__ dWk,                        // dW + k*B*L
    const float* __restrict__ xs_t,                       // xs + (k+1)*B*D
    const float* __restrict__ projW, const float* __restrict__ proj_b,
    const float* __restrict__ noise_std, float* __restrict__ acc) {
  __shared__ bf16  s_in[32][L_ + C_];   // [z | ctx]
  __shared__ bf16  s_a1[32][H_];
  __shared__ bf16  s_a2[32][H_];
  __shared__ float s_fo[32][L_];
  __shared__ float s_ho[32][L_];
  __shared__ float s_zn[32][L_];

  int lane = threadIdx.x & 31, w = threadIdx.x >> 5;
  int rows0 = blockIdx.x * 32;
  float dt = ts[k + 1] - ts[k];

  __builtin_prefetch(ctx_row + (size_t)rows0 * C_, 0, 1);  // global_prefetch_b8

  // stage inputs
  for (int i = threadIdx.x; i < 32 * L_; i += 64) {
    int r = i >> 5, c = i & (L_ - 1);
    s_in[r][c] = zbf_cur[(size_t)(rows0 + r) * L_ + c];
  }
  for (int i = threadIdx.x; i < 32 * C_; i += 64) {
    int r = i / C_, c = i % C_;
    s_in[r][L_ + c] = ctx_row[(size_t)(rows0 + r) * C_ + c];
  }
  __syncthreads();

  const int ar = w * 16;  // this wave's LDS row base
  int n = lane & 15, mb = (lane & 16) ? 8 : 0;

  // ---- f-net layer 1: [16x96] @ [96x256], softplus ----
  for (int nt = 0; nt < H_ / 16; ++nt) {
    v8f a = {};
    for (int kc = 0; kc < 3; ++kc)
      a = wmma_bf(frag_a(&s_in[0][0], L_ + C_, ar, kc * 32, 32),
                  frag_b(fW1, L_ + C_, kc * 32, nt * 16, 32), a);
    int col = nt * 16 + n;
    float bi = f_b1[col];
#pragma unroll
    for (int v = 0; v < 8; ++v) s_a1[ar + mb + v][col] = (bf16)splusf(a[v] + bi);
  }
  __syncthreads();
  // ---- f-net layer 2: [16x256] @ [256x256], softplus ----
  for (int nt = 0; nt < H_ / 16; ++nt) {
    v8f a = {};
    for (int kc = 0; kc < H_ / 32; ++kc)
      a = wmma_bf(frag_a(&s_a1[0][0], H_, ar, kc * 32, 32),
                  frag_b(fW2, H_, kc * 32, nt * 16, 32), a);
    int col = nt * 16 + n;
    float bi = f_b2[col];
#pragma unroll
    for (int v = 0; v < 8; ++v) s_a2[ar + mb + v][col] = (bf16)splusf(a[v] + bi);
  }
  __syncthreads();
  // ---- f-net layer 3: [16x256] @ [256x32] ----
  for (int nt = 0; nt < L_ / 16; ++nt) {
    v8f a = {};
    for (int kc = 0; kc < H_ / 32; ++kc)
      a = wmma_bf(frag_a(&s_a2[0][0], H_, ar, kc * 32, 32),
                  frag_b(fW3, H_, kc * 32, nt * 16, 32), a);
    int col = nt * 16 + n;
    float bi = f_b3[col];
#pragma unroll
    for (int v = 0; v < 8; ++v) s_fo[ar + mb + v][col] = a[v] + bi;
  }
  __syncthreads();
  // ---- h-net layer 1: [16x32] @ [32x256], softplus (reuse s_a1) ----
  for (int nt = 0; nt < H_ / 16; ++nt) {
    v8f a = {};
    a = wmma_bf(frag_a(&s_in[0][0], L_ + C_, ar, 0, 32),
                frag_b(hW1, L_, 0, nt * 16, 32), a);
    int col = nt * 16 + n;
    float bi = h_b1[col];
#pragma unroll
    for (int v = 0; v < 8; ++v) s_a1[ar + mb + v][col] = (bf16)splusf(a[v] + bi);
  }
  __syncthreads();
  // ---- h-net layer 2 ----
  for (int nt = 0; nt < H_ / 16; ++nt) {
    v8f a = {};
    for (int kc = 0; kc < H_ / 32; ++kc)
      a = wmma_bf(frag_a(&s_a1[0][0], H_, ar, kc * 32, 32),
                  frag_b(hW2, H_, kc * 32, nt * 16, 32), a);
    int col = nt * 16 + n;
    float bi = h_b2[col];
#pragma unroll
    for (int v = 0; v < 8; ++v) s_a2[ar + mb + v][col] = (bf16)splusf(a[v] + bi);
  }
  __syncthreads();
  // ---- h-net layer 3 ----
  for (int nt = 0; nt < L_ / 16; ++nt) {
    v8f a = {};
    for (int kc = 0; kc < H_ / 32; ++kc)
      a = wmma_bf(frag_a(&s_a2[0][0], H_, ar, kc * 32, 32),
                  frag_b(hW3, H_, kc * 32, nt * 16, 32), a);
    int col = nt * 16 + n;
    float bi = h_b3[col];
#pragma unroll
    for (int v = 0; v < 8; ++v) s_ho[ar + mb + v][col] = a[v] + bi;
  }
  __syncthreads();

  // ---- g_fn + u + Euler update + logqp_path accumulation ----
  float logr = 0.f;
  for (int i = lane; i < 16 * L_; i += 32) {
    int row = ar + (i >> 5), l = i & (L_ - 1);
    int b = rows0 + row;
    float zv = zf_cur[(size_t)b * L_ + l];
    float s = 0.f;
    for (int hh = 0; hh < H_; ++hh) {
      float a = splusf(zv * gW1[(size_t)l * H_ + hh] + gb1[(size_t)l * H_ + hh]);
      s += a * gW2[(size_t)l * H_ + hh];
    }
    float go = sigf(s + gb2[l]);
    float fv = s_fo[row][l], hv = s_ho[row][l];
    float u = (fv - hv) / go;
    float zn = zv + fv * dt + go * dWk[(size_t)b * L_ + l];
    zf_next [(size_t)b * L_ + l] = zn;
    zbf_next[(size_t)b * L_ + l] = (bf16)zn;
    s_zn[row][l] = zn;
    logr += 0.5f * u * u * dt;
  }
  logr = wave_sum(logr);
  if (lane == 0) atomicAdd(&acc[2], logr);
  __syncthreads();

  // ---- decode lp at t = k+1 ----
  float ns = noise_std[0], lns = __logf(ns);
  float lp = 0.f;
  for (int i = lane; i < 16 * D_; i += 32) {
    int row = ar + (i >> 4), d = i & (D_ - 1);
    int b = rows0 + row;
    float x = proj_b[d];
#pragma unroll 8
    for (int l = 0; l < L_; ++l) x += s_zn[row][l] * projW[d * L_ + l];
    float df = (xs_t[(size_t)b * D_ + d] - x) / ns;
    lp += -0.5f * df * df - lns - 0.91893853320467274f;
  }
  lp = wave_sum(lp);
  if (lane == 0) atomicAdd(&acc[0], lp);
}

__global__ void finalize_kernel(const float* __restrict__ acc, float* __restrict__ out) {
  if (threadIdx.x == 0 && blockIdx.x == 0) {
    out[0] = acc[0] / (float)B_;
    out[1] = (acc[1] + acc[2]) / (float)B_;
  }
}

// ---------------- host launcher ----------------

extern "C" void kernel_launch(void* const* d_in, const int* in_sizes, int n_in,
                              void* d_out, int out_size, void* d_ws, size_t ws_size,
                              hipStream_t stream) {
  const float* xs    = (const float*)d_in[0];
  const float* ts    = (const float*)d_in[1];
  const float* nstd  = (const float*)d_in[2];
  const float* eps0  = (const float*)d_in[3];
  const float* dW    = (const float*)d_in[4];
  const float* Wih   = (const float*)d_in[5];
  const float* Whh   = (const float*)d_in[6];
  const float* bih   = (const float*)d_in[7];
  const float* bhh   = (const float*)d_in[8];
  const float* encW  = (const float*)d_in[9];
  const float* enc_b = (const float*)d_in[10];
  const float* qz0W  = (const float*)d_in[11];
  const float* qz0b  = (const float*)d_in[12];
  const float* fW1 = (const float*)d_in[13]; const float* fb1 = (const float*)d_in[14];
  const float* fW2 = (const float*)d_in[15]; const float* fb2 = (const float*)d_in[16];
  const float* fW3 = (const float*)d_in[17]; const float* fb3 = (const float*)d_in[18];
  const float* hW1 = (const float*)d_in[19]; const float* hb1 = (const float*)d_in[20];
  const float* hW2 = (const float*)d_in[21]; const float* hb2 = (const float*)d_in[22];
  const float* hW3 = (const float*)d_in[23]; const float* hb3 = (const float*)d_in[24];
  const float* gW1 = (const float*)d_in[25]; const float* gb1 = (const float*)d_in[26];
  const float* gW2 = (const float*)d_in[27]; const float* gb2 = (const float*)d_in[28];
  const float* prW = (const float*)d_in[29]; const float* prb = (const float*)d_in[30];
  const float* pzm = (const float*)d_in[31]; const float* pzl = (const float*)d_in[32];

  char* w = (char*)d_ws;
  size_t off = 0;
  auto take = [&](size_t bytes) {
    void* p = w + off;
    off = (off + bytes + 255) & ~(size_t)255;
    return p;
  };
  bf16* xs_bf  = (bf16*)take((size_t)T_ * B_ * D_ * 2);
  bf16* Wih_bf = (bf16*)take((size_t)3 * H_ * D_ * 2);
  bf16* Whh_bf = (bf16*)take((size_t)3 * H_ * H_ * 2);
  bf16* enc_bf = (bf16*)take((size_t)C_ * H_ * 2);
  bf16* fW1_bf = (bf16*)take((size_t)H_ * (L_ + C_) * 2);
  bf16* fW2_bf = (bf16*)take((size_t)H_ * H_ * 2);
  bf16* fW3_bf = (bf16*)take((size_t)L_ * H_ * 2);
  bf16* hW1_bf = (bf16*)take((size_t)H_ * L_ * 2);
  bf16* hW2_bf = (bf16*)take((size_t)H_ * H_ * 2);
  bf16* hW3_bf = (bf16*)take((size_t)L_ * H_ * 2);
  bf16* ctx_bf = (bf16*)take((size_t)T_ * B_ * C_ * 2);
  float* hf0 = (float*)take((size_t)B_ * H_ * 4);
  float* hf1 = (float*)take((size_t)B_ * H_ * 4);
  bf16* hb0 = (bf16*)take((size_t)B_ * H_ * 2);
  bf16* hb1_ = (bf16*)take((size_t)B_ * H_ * 2);
  float* zf0 = (float*)take((size_t)B_ * L_ * 4);
  float* zf1 = (float*)take((size_t)B_ * L_ * 4);
  bf16* zb0 = (bf16*)take((size_t)B_ * L_ * 2);
  bf16* zb1 = (bf16*)take((size_t)B_ * L_ * 2);
  float* acc = (float*)take(4 * sizeof(float));

  float* hf[2] = {hf0, hf1};
  bf16*  hb[2] = {hb0, hb1_};
  float* zf[2] = {zf0, zf1};
  bf16*  zb[2] = {zb0, zb1};

  auto cvt = [&](const float* s, bf16* d, int n) {
    cvt_bf16_kernel<<<(n + 255) / 256, 256, 0, stream>>>(s, d, n);
  };
  cvt(xs, xs_bf, T_ * B_ * D_);
  cvt(Wih, Wih_bf, 3 * H_ * D_);
  cvt(Whh, Whh_bf, 3 * H_ * H_);
  cvt(encW, enc_bf, C_ * H_);
  cvt(fW1, fW1_bf, H_ * (L_ + C_));
  cvt(fW2, fW2_bf, H_ * H_);
  cvt(fW3, fW3_bf, L_ * H_);
  cvt(hW1, hW1_bf, H_ * L_);
  cvt(hW2, hW2_bf, H_ * H_);
  cvt(hW3, hW3_bf, L_ * H_);
  zero_f32_kernel<<<(B_ * H_ + 255) / 256, 256, 0, stream>>>(hf[0], B_ * H_);
  zero_bf16_kernel<<<(B_ * H_ + 255) / 256, 256, 0, stream>>>(hb[0], B_ * H_);
  zero_f32_kernel<<<1, 32, 0, stream>>>(acc, 4);

  // backward GRU scan + context head (ctx stored at flipped index)
  for (int r = 0; r < T_; ++r) {
    int t = T_ - 1 - r;
    gru_step_kernel<<<dim3(B_ / 64, H_ / 64), 128, 0, stream>>>(
        xs_bf + (size_t)t * B_ * D_, hb[r & 1], hf[r & 1], hf[(r + 1) & 1],
        hb[(r + 1) & 1], Wih_bf, Whh_bf, bih, bhh);
    ctx_step_kernel<<<B_ / 64, 128, 0, stream>>>(
        hb[(r + 1) & 1], enc_bf, enc_b, ctx_bf + (size_t)t * B_ * C_);
  }

  // posterior init condition + KL + t=0 reconstruction term
  qz0_kernel<<<B_ / 128, 128, 0, stream>>>(ctx_bf, qz0W, qz0b, eps0, pzm, pzl,
                                           zf[0], zb[0], xs, prW, prb, nstd, acc);

  // Euler-Maruyama scan; ts is uniform so searchsorted(ts, ts[k], right) == k+1
  for (int k = 0; k < T_ - 1; ++k) {
    int idx = (k + 1 < T_ - 1) ? (k + 1) : (T_ - 1);
    sde_step_kernel<<<B_ / 32, 64, 0, stream>>>(
        k, ts, zf[k & 1], zb[k & 1], zf[(k + 1) & 1], zb[(k + 1) & 1],
        ctx_bf + (size_t)idx * B_ * C_, fW1_bf, fb1, fW2_bf, fb2, fW3_bf, fb3,
        hW1_bf, hb1, hW2_bf, hb2, hW3_bf, hb3, gW1, gb1, gW2, gb2,
        dW + (size_t)k * B_ * L_, xs + (size_t)(k + 1) * B_ * D_, prW, prb, nstd,
        acc);
  }

  finalize_kernel<<<1, 32, 0, stream>>>(acc, (float*)d_out);
}
